// LongFormer_25615184953532
// MI455X (gfx1250) — matrix operand order repaired and checked by reading, not compiled, not run on hardware
//
#include <hip/hip_runtime.h>
#include <hip/hip_bf16.h>

// ---------------------------------------------------------------------------
// Longformer block for MI455X (gfx1250, wave32, WMMA + async/TDM staging).
// All matmuls run through v_wmma_f32_16x16x32_bf16 (f32 accum, bf16 operands).
// Inputs are converted f32->bf16 once; GEMM tile staging is then pure byte
// movement: per-lane GLOBAL_LOAD_ASYNC_TO_LDS_B128 for A tiles and a Tensor
// Data Mover descriptor load for B tiles, double buffered with
// s_wait_asynccnt / s_wait_tensorcnt.
// ---------------------------------------------------------------------------

#define SEQL 2048
#define EMB  512
#define NH   8
#define FEAT 64
#define WWIN 32
#define DILA 4
// keys per query = 2*WWIN+1 = 65; window halo = WWIN*DILA = 128

typedef __bf16 bf16_t;
typedef __attribute__((ext_vector_type(16))) __bf16 v16bf;
typedef __attribute__((ext_vector_type(8)))  float  v8f;
typedef unsigned int u32x4 __attribute__((ext_vector_type(4)));
typedef int          i32x4 __attribute__((ext_vector_type(4)));
typedef int          i32x8 __attribute__((ext_vector_type(8)));
typedef int          v4i_vs __attribute__((vector_size(16)));

// ---------------------------------------------------------------------------
// CDNA5 data-movement wrappers (guarded; fall back to synchronous copies).
// ---------------------------------------------------------------------------
__device__ inline void async_copy16(const bf16_t* g, bf16_t* l) {
#if __has_builtin(__builtin_amdgcn_global_load_async_to_lds_b128)
    __builtin_amdgcn_global_load_async_to_lds_b128(
        (__attribute__((address_space(1))) v4i_vs*)g,
        (__attribute__((address_space(3))) v4i_vs*)l, 0, 0);
#else
    *(uint4*)l = *(const uint4*)g;
#endif
}

__device__ inline void wait_async_tensor() {
#if __has_builtin(__builtin_amdgcn_s_wait_asynccnt)
    __builtin_amdgcn_s_wait_asynccnt(0);
#else
    asm volatile("s_wait_asynccnt 0x0" ::: "memory");
#endif
#if __has_builtin(__builtin_amdgcn_s_wait_tensorcnt)
    __builtin_amdgcn_s_wait_tensorcnt(0);
#else
    asm volatile("s_wait_tensorcnt 0x0" ::: "memory");
#endif
}

#if __has_builtin(__builtin_amdgcn_tensor_load_to_lds)
#define HAVE_TDM 1
#else
#define HAVE_TDM 0
#endif

#if HAVE_TDM
// 2D TDM load: `rows` x `row_len` bf16 tile from a row-major tensor with
// row stride `row_stride` (elements) into contiguous LDS at lds_off.
// D# layout per CDNA5 ISA.
// This toolchain exposes the 6-arg builtin form:
//   (u32x4 g0, i32x8 g1, i32x4 g2, i32x4 g3, i32x8 extra, int cpol)
__device__ inline void tdm_load_tile_2d(const bf16_t* gsrc, unsigned lds_off,
                                        int rows, int row_len,
                                        int row_stride) {
    const unsigned long long ga = (unsigned long long)(size_t)gsrc;
    const u32x4 g0 = {
        1u,                                   // count=1, user descriptor
        lds_off,                              // lds_addr (bytes)
        (unsigned)(ga & 0xffffffffu),         // global_addr[31:0]
        (unsigned)((ga >> 32) & 0x1ffffffu) | (2u << 30)  // addr[56:32], type=2
    };
    const unsigned d0 = (unsigned)row_stride;  // tensor_dim0 (elements)
    const unsigned d1 = 1u << 20;              // tensor_dim1: large (in-bounds)
    const i32x8 g1 = {
        (int)(1u << 16),                       // data_size=1 (2 bytes/elem)
        (int)((d0 & 0xffffu) << 16),           // tensor_dim0[15:0]
        (int)(((d0 >> 16) & 0xffffu) | ((d1 & 0xffffu) << 16)),
        (int)(((d1 >> 16) & 0xffffu) | ((unsigned)row_len << 16)),  // tile_dim0
        (int)(unsigned)rows,                   // tile_dim1 (tile_dim2 = 0)
        (int)(unsigned)row_stride,             // tensor_dim0_stride[31:0]
        0,                                     // stride hi, dim1_stride lo
        0
    };
    const i32x4 gz  = {0, 0, 0, 0};
    const i32x8 gz8 = {0, 0, 0, 0, 0, 0, 0, 0};
    __builtin_amdgcn_tensor_load_to_lds(g0, g1, gz, gz, gz8, 0);
}
#endif

// ---------------------------------------------------------------------------
// Fragment builders in contiguous k order (lane half h, element e -> k = 16h + e; A and B alike).
// ---------------------------------------------------------------------------
__device__ inline v16bf load_a_frag(const bf16_t* base, int stride, int m0,
                                    int k0, int lane) {
    const int half = lane >> 4;
    const int row  = m0 + (lane & 15);
    const bf16_t* p = base + row * stride + k0;
    v16bf a;
#pragma unroll
    for (int e = 0; e < 16; ++e) {
        int k = e + (half ? 16 : 0);
        a[e] = p[k];
    }
    return a;
}

// B: 32x16 bf16 stored [K][N]; lanes 0-15: K = k0+e, lanes 16-31: K = k0+16+e.
__device__ inline v16bf load_b_frag(const bf16_t* base, int stride, int k0,
                                    int n0, int lane) {
    const int half = lane >> 4;
    const int n    = n0 + (lane & 15);
    v16bf b;
#pragma unroll
    for (int e = 0; e < 16; ++e) {
        int k = k0 + e + (half ? 16 : 0);
        b[e] = base[k * stride + n];
    }
    return b;
}

// B built from a row-major [N][K] source (implicit transpose, contiguous).
__device__ inline v16bf load_bT_frag(const bf16_t* base, int stride, int k0,
                                     int n0, int lane) {
    const int half = lane >> 4;
    const int n    = n0 + (lane & 15);
    const bf16_t* p = base + n * stride + k0 + (half ? 16 : 0);
    v16bf b;
#pragma unroll
    for (int e = 0; e < 16; ++e) b[e] = p[e];
    return b;
}

__device__ inline v8f wmma_bf16(v16bf a, v16bf b, v8f c) {
    return __builtin_amdgcn_wmma_f32_16x16x32_bf16(
        false, a, false, b, (short)0, c, false, false);
}

// ---------------------------------------------------------------------------
// Elementwise f32 -> bf16 conversion (one streaming pass over x + weights).
// ---------------------------------------------------------------------------
__global__ __launch_bounds__(256) void f32_to_bf16_kernel(
    const float* __restrict__ s, bf16_t* __restrict__ d, int n) {
    const int i = (blockIdx.x * 256 + threadIdx.x) * 4;
    if (i < n) {
        const float4 v = *(const float4*)&s[i];
        bf16_t* o = &d[i];
        o[0] = (bf16_t)v.x; o[1] = (bf16_t)v.y;
        o[2] = (bf16_t)v.z; o[3] = (bf16_t)v.w;
    }
}

// ---------------------------------------------------------------------------
// Tiled bf16 GEMM: C(MxN) = A(MxK) * B(KxN). 128 threads = 4 waves,
// 64x64 tile, K-step 32, double-buffered async/TDM staging.
// MODE 0: store bf16 per-head layout out[h][row][feat]      (QKV projection)
// MODE 1: relu(C + bias) -> bf16 outB (row-major)           (FFN layer 1)
// MODE 2: resid + relu(C + bias) -> f32 outF                (FFN layer 2)
// ---------------------------------------------------------------------------
template <int MODE>
__global__ __launch_bounds__(128) void gemm_wmma_kernel(
    const bf16_t* __restrict__ A, const bf16_t* __restrict__ Bw,
    const float* __restrict__ bias, const float* __restrict__ resid,
    float* __restrict__ outF, bf16_t* __restrict__ outB,
    int M, int N, int K) {
    __shared__ __align__(16) bf16_t As[2][64 * 32];
    __shared__ __align__(16) bf16_t Bs[2][32 * 64];

    const int tid  = threadIdx.x;
    const int lane = tid & 31;
    const int wave = tid >> 5;
    const int bm   = blockIdx.y * 64;
    const int bn   = blockIdx.x * 64;
    const int KT   = K >> 5;

    v8f acc[4] = {};

    // Tile staging: A = 64x32 (256 16B chunks, 2/thread) async to LDS;
    // B = 32x64 via one TDM descriptor (wave-level, thread 0).
    auto stage = [&](int buf, int kt) {
        const int k0 = kt << 5;
#pragma unroll
        for (int i = 0; i < 2; ++i) {
            const int chunk = tid + i * 128;
            const int r = chunk >> 2;
            const int c = (chunk & 3) * 8;
            async_copy16(&A[(size_t)(bm + r) * K + k0 + c],
                         &As[buf][r * 32 + c]);
        }
#if HAVE_TDM
        if (tid == 0) {
            const unsigned lds_off = (unsigned)(size_t)(void*)&Bs[buf][0];
            tdm_load_tile_2d(&Bw[(size_t)k0 * N + bn], lds_off, 32, 64, N);
        }
#else
#pragma unroll
        for (int i = 0; i < 2; ++i) {
            const int chunk = tid + i * 128;
            const int r = chunk >> 3;
            const int c = (chunk & 7) * 8;
            async_copy16(&Bw[(size_t)(k0 + r) * N + bn + c],
                         &Bs[buf][r * 64 + c]);
        }
#endif
    };

    stage(0, 0);
    wait_async_tensor();
    __syncthreads();

    for (int kt = 0; kt < KT; ++kt) {
        const int cur = kt & 1;
        if (kt + 1 < KT) stage(1 - cur, kt + 1);

        const v16bf a = load_a_frag(As[cur], 32, wave * 16, 0, lane);
#pragma unroll
        for (int nf = 0; nf < 4; ++nf) {
            const v16bf b = load_b_frag(Bs[cur], 64, 0, nf * 16, lane);
            acc[nf] = wmma_bf16(a, b, acc[nf]);
        }

        wait_async_tensor();
        __syncthreads();
    }

    // Epilogue. C frag layout: VGPR rr -> M = rr (lanes 0-15) / rr+8 (16-31).
    const int mlo = bm + wave * 16 + ((lane < 16) ? 0 : 8);
#pragma unroll
    for (int nf = 0; nf < 4; ++nf) {
        const int col = bn + nf * 16 + (lane & 15);
#pragma unroll
        for (int rr = 0; rr < 8; ++rr) {
            const int row = mlo + rr;
            const float cv = acc[nf][rr];
            if (MODE == 0) {
                outB[(size_t)(col >> 6) * (SEQL * FEAT) + (size_t)row * FEAT +
                     (col & 63)] = (bf16_t)cv;
            } else if (MODE == 1) {
                float v = cv + bias[col];
                v = v > 0.f ? v : 0.f;
                outB[(size_t)row * N + col] = (bf16_t)v;
            } else {
                float v = cv + bias[col];
                v = v > 0.f ? v : 0.f;
                outF[(size_t)row * N + col] =
                    resid[(size_t)row * N + col] + v;
            }
        }
    }
}

// ---------------------------------------------------------------------------
// Dilated windowed attention. One wave32 per workgroup handles one
// (head, residue, 16-query tile): queries s_i = r + 64*t + 4*i, i = 0..15.
// Key windows union to 80 stride-4 positions p_m = base + 4*m, m = 0..79,
// base = r + 64*t - 128; query i uses m = i..i+64.
// scores = Q(16x64) @ Ksub^T (5x2 WMMA); softmax over 65-band then /sqrt(64)
// (the reference divides AFTER softmax); ctx = bandP(16x96) @ Vsub (4x3 WMMA).
// ---------------------------------------------------------------------------
__global__ __launch_bounds__(32) void attn_kernel(
    const bf16_t* __restrict__ q, const bf16_t* __restrict__ k,
    const bf16_t* __restrict__ v, float* __restrict__ x_attn,
    bf16_t* __restrict__ x_attn_bf) {
    __shared__ __align__(16) bf16_t Qs[16 * 64];
    __shared__ __align__(16) bf16_t Ks[80 * 64];
    __shared__ __align__(16) bf16_t Vs[96 * 64];
    __shared__ float  Ss[16 * 80];
    __shared__ bf16_t Ps[16 * 96];

    const int wg   = blockIdx.x;    // 0..1023
    const int h    = wg >> 7;       // head
    const int r    = (wg >> 5) & 3; // residue class (s mod 4)
    const int t    = wg & 31;       // tile within residue class
    const int lane = threadIdx.x;
    const int base = r + 64 * t - 128;

    const bf16_t* qh = q + (size_t)h * SEQL * FEAT;
    const bf16_t* kh = k + (size_t)h * SEQL * FEAT;
    const bf16_t* vh = v + (size_t)h * SEQL * FEAT;

    // Q tile: 16 rows x 64 bf16 in 16B chunks, async to LDS.
    for (int c = lane; c < 128; c += 32) {
        const int row = c >> 3, cc = (c & 7) * 8;
        const int s = r + 64 * t + 4 * row;
        async_copy16(&qh[(size_t)s * FEAT + cc], &Qs[row * 64 + cc]);
    }
    // K rows m=0..79; zero-fill outside sequence (window padding).
    const uint4 zero16 = {0u, 0u, 0u, 0u};
    for (int c = lane; c < 640; c += 32) {
        const int m = c >> 3, cc = (c & 7) * 8;
        const int p = base + 4 * m;
        if (p >= 0 && p < SEQL)
            async_copy16(&kh[(size_t)p * FEAT + cc], &Ks[m * 64 + cc]);
        else
            *(uint4*)&Ks[m * 64 + cc] = zero16;
    }
    // V rows m=0..79 (+ rows 80..95 zero pad for the K=96 contraction).
    for (int c = lane; c < 768; c += 32) {
        const int m = c >> 3, cc = (c & 7) * 8;
        const int p = base + 4 * m;
        if (m < 80 && p >= 0 && p < SEQL)
            async_copy16(&vh[(size_t)p * FEAT + cc], &Vs[m * 64 + cc]);
        else
            *(uint4*)&Vs[m * 64 + cc] = zero16;
    }
    wait_async_tensor();
    __syncthreads();

    // Scores: S(16x80) = Q @ Ksub^T (contraction over FEAT=64, 2 k-steps).
#pragma unroll
    for (int nt = 0; nt < 5; ++nt) {
        v8f c = {};
#pragma unroll
        for (int kb = 0; kb < 2; ++kb) {
            const v16bf a = load_a_frag(Qs, 64, 0, kb * 32, lane);
            const v16bf b = load_bT_frag(Ks, 64, kb * 32, nt * 16, lane);
            c = wmma_bf16(a, b, c);
        }
        const int nn = nt * 16 + (lane & 15);
        const int mb = (lane < 16) ? 0 : 8;
#pragma unroll
        for (int rr = 0; rr < 8; ++rr) Ss[(mb + rr) * 80 + nn] = c[rr];
    }
    __syncthreads();

    // Softmax over the 65-wide band of row i (cols i..i+64), then /sqrt(64).
    if (lane < 16) {
        const int i = lane;
        float mx = -1e30f;
        for (int j = 0; j <= 64; ++j)
            mx = fmaxf(mx, Ss[i * 80 + i + j]);
        float sum = 0.f;
        for (int j = 0; j <= 64; ++j)
            sum += __expf(Ss[i * 80 + i + j] - mx);
        const float inv = 1.f / (sum * 8.0f);  // sqrt(FEAT) = 8
        for (int m = 0; m < 96; ++m) {
            const int j = m - i;
            float pv = 0.f;
            if (j >= 0 && j <= 64) pv = __expf(Ss[i * 80 + m] - mx) * inv;
            Ps[i * 96 + m] = (bf16_t)pv;
        }
    }
    __syncthreads();

    // Context: ctx(16x64) = bandP(16x96) @ Vsub(96x64), 3 k-steps.
#pragma unroll
    for (int nt = 0; nt < 4; ++nt) {
        v8f c = {};
#pragma unroll
        for (int kb = 0; kb < 3; ++kb) {
            const v16bf a = load_a_frag(Ps, 96, 0, kb * 32, lane);
            const v16bf b = load_b_frag(Vs, 64, kb * 32, nt * 16, lane);
            c = wmma_bf16(a, b, c);
        }
        const int n  = nt * 16 + (lane & 15);
        const int mb = (lane < 16) ? 0 : 8;
#pragma unroll
        for (int rr = 0; rr < 8; ++rr) {
            const int i = mb + rr;
            const int s = r + 64 * t + 4 * i;
            const float cv = c[rr];
            x_attn[(size_t)s * EMB + h * FEAT + n]    = cv;
            x_attn_bf[(size_t)s * EMB + h * FEAT + n] = (bf16_t)cv;
        }
    }
}

// ---------------------------------------------------------------------------
// Host-side orchestration (all on `stream`, graph-capture safe).
// Workspace layout (bytes):
//   [0.0M, 2.0M)  xb   bf16          [2.0M, 2.5M)  Wq  bf16
//   [2.5M, 3.0M)  Wk   bf16          [3.0M, 3.5M)  Wv  bf16
//   [3.5M, 4.0M)  W1   bf16          [4.0M, 4.5M)  W2  bf16
//   [4.5M, 6.5M)  q    bf16          [6.5M, 8.5M)  k   bf16
//   [8.5M,10.5M)  v    bf16          [10.5M,14.5M) x_attn f32
//   [14.5M,16.5M) x_attn bf16        [16.5M,18.5M) f   bf16
// ---------------------------------------------------------------------------
extern "C" void kernel_launch(void* const* d_in, const int* in_sizes, int n_in,
                              void* d_out, int out_size, void* d_ws,
                              size_t ws_size, hipStream_t stream) {
    const float* x    = (const float*)d_in[0];
    const float* Wq   = (const float*)d_in[1];
    const float* Wk   = (const float*)d_in[2];
    const float* Wv   = (const float*)d_in[3];
    const float* Wff1 = (const float*)d_in[4];
    const float* bff1 = (const float*)d_in[5];
    const float* Wff2 = (const float*)d_in[6];
    const float* bff2 = (const float*)d_in[7];

    char* ws = (char*)d_ws;
    const size_t KB512 = 512ull * 1024ull;
    bf16_t* xb   = (bf16_t*)(ws + 0 * KB512);
    bf16_t* Wqb  = (bf16_t*)(ws + 4 * KB512);
    bf16_t* Wkb  = (bf16_t*)(ws + 5 * KB512);
    bf16_t* Wvb  = (bf16_t*)(ws + 6 * KB512);
    bf16_t* W1b  = (bf16_t*)(ws + 7 * KB512);
    bf16_t* W2b  = (bf16_t*)(ws + 8 * KB512);
    bf16_t* qb   = (bf16_t*)(ws + 9 * KB512);
    bf16_t* kb   = (bf16_t*)(ws + 13 * KB512);
    bf16_t* vb   = (bf16_t*)(ws + 17 * KB512);
    float*  xatt = (float*)(ws + 21 * KB512);
    bf16_t* xatb = (bf16_t*)(ws + 29 * KB512);
    bf16_t* fb   = (bf16_t*)(ws + 33 * KB512);

    // f32 -> bf16 conversions (x + 5 weight matrices).
    const int NX = SEQL * EMB, NW = EMB * EMB;
    f32_to_bf16_kernel<<<NX / 1024, 256, 0, stream>>>(x, xb, NX);
    f32_to_bf16_kernel<<<NW / 1024, 256, 0, stream>>>(Wq, Wqb, NW);
    f32_to_bf16_kernel<<<NW / 1024, 256, 0, stream>>>(Wk, Wkb, NW);
    f32_to_bf16_kernel<<<NW / 1024, 256, 0, stream>>>(Wv, Wvb, NW);
    f32_to_bf16_kernel<<<NW / 1024, 256, 0, stream>>>(Wff1, W1b, NW);
    f32_to_bf16_kernel<<<NW / 1024, 256, 0, stream>>>(Wff2, W2b, NW);

    const dim3 gGemm(EMB / 64, SEQL / 64);  // 8 x 32
    const dim3 bGemm(128);

    // QKV projections -> bf16 per-head layout.
    gemm_wmma_kernel<0><<<gGemm, bGemm, 0, stream>>>(
        xb, Wqb, nullptr, nullptr, nullptr, qb, SEQL, EMB, EMB);
    gemm_wmma_kernel<0><<<gGemm, bGemm, 0, stream>>>(
        xb, Wkb, nullptr, nullptr, nullptr, kb, SEQL, EMB, EMB);
    gemm_wmma_kernel<0><<<gGemm, bGemm, 0, stream>>>(
        xb, Wvb, nullptr, nullptr, nullptr, vb, SEQL, EMB, EMB);

    // Dilated windowed attention: 8 heads x 4 residues x 32 tiles.
    attn_kernel<<<dim3(NH * 4 * 32), dim3(32), 0, stream>>>(qb, kb, vb, xatt,
                                                            xatb);

    // FFN1: f = relu(x_attn @ Wff1 + b1) -> bf16
    gemm_wmma_kernel<1><<<gGemm, bGemm, 0, stream>>>(
        xatb, W1b, bff1, nullptr, nullptr, fb, SEQL, EMB, EMB);
    // FFN2 + residual: out = x_attn + relu(f @ Wff2 + b2)
    gemm_wmma_kernel<2><<<gGemm, bGemm, 0, stream>>>(
        fb, W2b, bff2, xatt, (float*)d_out, nullptr, SEQL, EMB, EMB);
}
